// GTLUTProduct_23871428231429
// MI455X (gfx1250) — matrix operand order repaired and checked by reading, not compiled, not run on hardware
//
#include <hip/hip_runtime.h>
#include <hip/hip_bf16.h>

typedef __attribute__((ext_vector_type(16))) _Float16 v16h;
typedef __attribute__((ext_vector_type(8)))  float    v8f;

#define Bn 4
#define Ln 256
#define Fh 64        // width of each of the three feature sections
#define Dn 8
#define Kn 6
#define DK 48        // D*K
#define An 64        // 1<<K
#define On 64
#define DA 512       // D*A  (K dim of the final GEMM)

// workspace layout (in floats)
#define G1_OFF 0
#define G2_OFF (Bn*Ln*DK)
#define G3_OFF (2*Bn*Ln*DK)
#define SS_OFF (2*Bn*Ln*DK + Ln*DK)

// ---------------------------------------------------------------------------
// Kernel A: separable projections.
//   G1[b,l,dk] = input_1[b,l,:64] . anchors[dk, 0:64]
//   G2[b,l,dk] = input_2[b,l,:64] . anchors[dk, 64:128]
//   G3[dl,dk]  = pos_emb[dl,:64]  . anchors[dk, 128:192]
// ---------------------------------------------------------------------------
__global__ __launch_bounds__(256) void k_proj(const float* __restrict__ in1,
                                              const float* __restrict__ in2,
                                              const float* __restrict__ pos,
                                              const float* __restrict__ anchors,
                                              float* __restrict__ ws) {
  int idx = blockIdx.x * blockDim.x + threadIdx.x;
  const int nG = Bn * Ln * DK;
  const int total = 2 * nG + Ln * DK;
  if (idx >= total) return;

  const float* src;
  float* dst;
  int f0, row, col;
  if (idx < nG) {
    src = in1; f0 = 0; row = idx / DK; col = idx % DK; dst = ws + G1_OFF + idx;
  } else if (idx < 2 * nG) {
    int t = idx - nG;
    src = in2; f0 = 64; row = t / DK; col = t % DK; dst = ws + G2_OFF + t;
  } else {
    int t = idx - 2 * nG;
    src = pos; f0 = 128; row = t / DK; col = t % DK; dst = ws + G3_OFF + t;
  }
  const float* x = src + row * Fh;
  const float* a = anchors + col * 192 + f0;   // anchors[d,k,f] flat, col = d*6+k
  float s = 0.f;
#pragma unroll 8
  for (int f = 0; f < Fh; ++f) s += x[f] * a[f];
  *dst = s;
}

// ---------------------------------------------------------------------------
// Kernel B: Ssum[b,j,d*64+a] = sum_{i<j} prod_k sel(bit_k(a), sig(b,i,j,d,k))
// One block per (b,j). 256 threads = 8 detectors x 32 a-pair groups.
// Sigmoids for a chunk of 32 i-values are staged in LDS and broadcast.
// ---------------------------------------------------------------------------
__global__ __launch_bounds__(256) void k_ssum(const float* __restrict__ ws_g,
                                              float* __restrict__ ssum) {
  const int j = blockIdx.x;
  const int b = blockIdx.y;
  const int tid = threadIdx.x;

  __shared__ float sig[32][DK];
  __shared__ float g2j[DK];

  const float* G1 = ws_g + G1_OFF + (b * Ln) * DK;
  const float* G2 = ws_g + G2_OFF + (b * Ln + j) * DK;
  const float* G3 = ws_g + G3_OFF;

  if (tid < DK) g2j[tid] = G2[tid];
  __syncthreads();

  const int d  = tid >> 5;          // detector 0..7
  const int a0 = (tid & 31) << 1;   // even LUT address; thread owns {a0, a0+1}
  float acc0 = 0.f, acc1 = 0.f;

  for (int i0 = 0; i0 < j; i0 += 32) {
    const int nI = min(32, j - i0);
    // stage sigmoids: sig[il][c] = sigmoid(G1[i]+G2[j]+G3[j-i]) at column c
    for (int s = tid; s < nI * DK; s += 256) {
      int il = s / DK, c = s % DK;
      int i = i0 + il;
      float x = G1[i * DK + c] + g2j[c] + G3[(j - i) * DK + c];
      sig[il][c] = 1.f / (1.f + __expf(-x));
    }
    __syncthreads();

    for (int il = 0; il < nI; ++il) {
      const float* sd = &sig[il][d * Kn];
      float common = 1.f;
#pragma unroll
      for (int k = 1; k < Kn; ++k) {
        float sk = sd[k];
        common *= ((a0 >> k) & 1) ? sk : (1.f - sk);
      }
      float s0 = sd[0];
      acc0 += common * (1.f - s0);  // address a0   (bit0 = 0)
      acc1 += common * s0;          // address a0+1 (bit0 = 1)
    }
    __syncthreads();
  }

  float* o = ssum + (b * Ln + j) * DA + d * An + a0;
  o[0] = acc0;
  o[1] = acc1;
}

// ---------------------------------------------------------------------------
// Kernel C: Out[1024,64] = Ssum[1024,512] x W[512,64] via v_wmma_f32_16x16x32_f16
// 64 blocks (16-row M tiles) x 4 waves (16-col N tiles), 16 K-chunks of 32.
// Fragment layouts per CDNA5 ISA 7.12.2 (wave32).
// ---------------------------------------------------------------------------
__global__ __launch_bounds__(128) void k_wmma(const float* __restrict__ ssum,
                                              const float* __restrict__ lutw,
                                              float* __restrict__ out) {
  const int tile  = blockIdx.x;          // M tile (16 rows of B*L = 1024)
  const int wave  = threadIdx.x >> 5;    // N tile (16 of O = 64)
  const int lane  = threadIdx.x & 31;
  const int row16 = lane & 15;
  const int hi    = lane >> 4;           // lane-half select
  const int mbase = tile * 16;

  v8f c = {};
  for (int kc = 0; kc < 16; ++kc) {
    v16h a, bm;
    // A fragment: lane holds row M=row16; half h -> K = kc*32 + hi*8 + (h&7) + (h>=8)*16
    const float* arow = ssum + (mbase + row16) * DA + kc * 32 + hi * 8;
#pragma unroll
    for (int h = 0; h < 16; ++h) {
      int koff = (h & 7) + ((h >= 8) ? 16 : 0);
      a[h] = (_Float16)arow[koff];
    }
    // B fragment: lane holds col N=row16; half h -> K = kc*32 + hi*8 + (h&7) + (h>=8)*16
    const int kbaseB = kc * 32 + hi * 8;
#pragma unroll
    for (int h = 0; h < 16; ++h) {
      bm[h] = (_Float16)lutw[(kbaseB + (h & 7) + ((h >= 8) ? 16 : 0)) * On + wave * 16 + row16];
    }
    c = __builtin_amdgcn_wmma_f32_16x16x32_f16(
        /*neg_a=*/false, a, /*neg_b=*/false, bm,
        /*c_mod=*/(short)0, c, /*reuse_a=*/false, /*reuse_b=*/false);
  }

  // C/D layout: VGPR r, lanes 0-15 -> M=r, lanes 16-31 -> M=r+8
  const int mrow = mbase + hi * 8;
#pragma unroll
  for (int r = 0; r < 8; ++r) {
    out[(mrow + r) * On + wave * 16 + row16] = c[r];
  }
}

// ---------------------------------------------------------------------------
extern "C" void kernel_launch(void* const* d_in, const int* in_sizes, int n_in,
                              void* d_out, int out_size, void* d_ws, size_t ws_size,
                              hipStream_t stream) {
  (void)in_sizes; (void)n_in; (void)out_size; (void)ws_size;

  const float* in1     = (const float*)d_in[0];  // [B,L,64]
  const float* in2     = (const float*)d_in[1];  // [B,L,64]
  const float* pos     = (const float*)d_in[2];  // [L,64]
  const float* anchors = (const float*)d_in[3];  // [D,K,192]
  const float* lutw    = (const float*)d_in[4];  // [D,A,O] == [512,64]
  float* out = (float*)d_out;                    // [B,L,O]
  float* ws  = (float*)d_ws;
  float* ssum = ws + SS_OFF;                     // [B*L, 512]

  const int totalA = 2 * Bn * Ln * DK + Ln * DK;
  k_proj<<<(totalA + 255) / 256, 256, 0, stream>>>(in1, in2, pos, anchors, ws);

  dim3 gridB(Ln, Bn);
  k_ssum<<<gridB, 256, 0, stream>>>(ws, ssum);

  k_wmma<<<(Bn * Ln) / 16, 128, 0, stream>>>(ssum, lutw, out);
}
